// GrafoNeuronal_25486335935216
// MI455X (gfx1250) — matrix operand order repaired and checked by reading, not compiled, not run on hardware
//
#include <hip/hip_runtime.h>
#include <math.h>

#define N_NODES 8192
#define F_IN    256
#define HDIM    128

typedef __attribute__((ext_vector_type(16))) __bf16 v16bf;
typedef __attribute__((ext_vector_type(8)))  float  v8f;
typedef __attribute__((ext_vector_type(4)))  float  v4f;
typedef __attribute__((ext_vector_type(4)))  int    v4i;

union BFrag {
    v16bf v;
    v4f   q[2];
};

// --------------------------------------------------------------------------
// Async global->LDS staging (CDNA5 GLOBAL_LOAD_ASYNC_TO_LDS_B128, ASYNCcnt)
// with synchronous fallback if the builtin is unavailable.
// Probe-learned signature: param 0 is `int4 __device__*` (addrspace(1), non-const).
// --------------------------------------------------------------------------
#if defined(__has_builtin)
#if __has_builtin(__builtin_amdgcn_global_load_async_to_lds_b128)
#define USE_ASYNC_LDS 1
#endif
#endif

typedef __attribute__((address_space(1))) v4i g_v4i;
typedef __attribute__((address_space(3))) v4i l_v4i;

__device__ __forceinline__ void copy16B_to_lds(const __bf16* src, __bf16* dst) {
#if defined(USE_ASYNC_LDS)
    __builtin_amdgcn_global_load_async_to_lds_b128(
        (g_v4i*)(uintptr_t)src,
        (l_v4i*)(uintptr_t)dst,
        0, 0);
#else
    *(v4f*)dst = *(const v4f*)src;
#endif
}

__device__ __forceinline__ void wait_async_lds() {
#if defined(USE_ASYNC_LDS)
#if __has_builtin(__builtin_amdgcn_s_wait_asynccnt)
    __builtin_amdgcn_s_wait_asynccnt(0);
#else
    asm volatile("s_wait_asynccnt 0x0" ::: "memory");
#endif
#endif
}

// Build a bf16 A-fragment (16x32 tile, this lane's 16 elements) from fp32 memory.
// Layout per CDNA5 ISA 7.12.2 (16-bit A 16x32): lane holds row M = lane%16;
// VGPRs 0-3: K = base..base+7, VGPRs 4-7: K = base+16..base+23,
// base = (lane<16 ? 0 : 8). Caller passes ap = &A[row][k0 + (lane>>4)*8].
// Non-temporal: A is streamed once per pass and exceeds L2; keep L2 for Bt/h1.
__device__ __forceinline__ v16bf cvt_a_frag(const float* __restrict__ ap) {
    v4f a0 = __builtin_nontemporal_load((const v4f*)(ap));
    v4f a1 = __builtin_nontemporal_load((const v4f*)(ap + 4));
    v4f a2 = __builtin_nontemporal_load((const v4f*)(ap + 16));
    v4f a3 = __builtin_nontemporal_load((const v4f*)(ap + 20));
    v16bf r;
#pragma unroll
    for (int j = 0; j < 4; ++j) {
        r[j]      = (__bf16)a0[j];
        r[4 + j]  = (__bf16)a1[j];
        r[8 + j]  = (__bf16)a2[j];
        r[12 + j] = (__bf16)a3[j];
    }
    return r;
}

// ---------------------------------------------------------------------------
// Kernel 1: transpose weights to bf16, K-contiguous per output channel.
// ---------------------------------------------------------------------------
__global__ void prep_weights(const float* __restrict__ W1, const float* __restrict__ W2,
                             __bf16* __restrict__ W1t, __bf16* __restrict__ W2t) {
    int i = blockIdx.x * blockDim.x + threadIdx.x;
    if (i < HDIM * F_IN) {
        int n = i / F_IN, k = i % F_IN;
        W1t[i] = (__bf16)W1[k * HDIM + n];
    } else {
        int j = i - HDIM * F_IN;
        if (j < HDIM * HDIM) {
            int n = j / HDIM, k = j % HDIM;
            W2t[j] = (__bf16)W2[k * HDIM + n];
        }
    }
}

// ---------------------------------------------------------------------------
// Kernel 2: small feature GEMM, output transposed bf16.
// BtOut[n][i] = sum_k Ain[i][k] * Wt[n][k]
// ---------------------------------------------------------------------------
__global__ void __launch_bounds__(256)
small_gemm(const float* __restrict__ Ain, const __bf16* __restrict__ Wt,
           __bf16* __restrict__ BtOut, int K) {
    const int lane = threadIdx.x & 31;
    const int w    = threadIdx.x >> 5;
    const int t    = blockIdx.x * 8 + w;
    const int it   = t >> 3;
    const int nt   = t & 7;
    const int m0   = it * 16;
    const int n0   = nt * 16;
    const int l16  = lane & 15;
    const int hi   = lane >> 4;

    const float*  arow = Ain + (size_t)(m0 + l16) * K;
    const __bf16* wrow = Wt  + (size_t)(n0 + l16) * K;

    v8f acc = {};
    for (int k0 = 0; k0 < K; k0 += 32) {
        v16bf af = cvt_a_frag(arow + k0 + hi * 8);
        BFrag bfr;
        const __bf16* bp = wrow + k0 + hi * 8;
        bfr.q[0] = *(const v4f*)bp;
        bfr.q[1] = *(const v4f*)(bp + 16);
        acc = __builtin_amdgcn_wmma_f32_16x16x32_bf16(
                  false, af, false, bfr.v, (short)0, acc, false, false);
    }

    union { __bf16 h[8]; v4f q; } outp;
#pragma unroll
    for (int r = 0; r < 8; ++r) outp.h[r] = (__bf16)acc[r];
    *(v4f*)&BtOut[(size_t)(n0 + l16) * N_NODES + m0 + hi * 8] = outp.q;
}

// ---------------------------------------------------------------------------
// Kernel 3: big GEMM  D = act(A @ Bt^T + bias)
// Double-buffered LDS, async global->LDS B staging, K-chunk = 64 (2 WMMA steps).
// Workgroup: 128 rows x 128 cols, 8 waves (each 16 rows x 8 WMMA tiles).
// mode 0: Hout = relu(acc + bias);  mode 1: Out = sigmoid(relu(...)@Wf + Bf)
// ---------------------------------------------------------------------------
#define BT_LDS_STRIDE 40                    // bf16; 80B rows -> 16B-aligned, bank-clean
#define PLANE (HDIM * BT_LDS_STRIDE)        // one 32-K half-plane
#define KCHUNK 64
#define NITER (N_NODES / KCHUNK)            // 128

__global__ void __launch_bounds__(256)
big_gemm(const float* __restrict__ A, const __bf16* __restrict__ Bt,
         const float* __restrict__ bias, float* __restrict__ Hout,
         const float* __restrict__ Wf, const float* __restrict__ Bf,
         float* __restrict__ Out, int mode) {
    __shared__ __attribute__((aligned(16))) __bf16 bs[2][2][PLANE];

    const int lane = threadIdx.x & 31;
    const int w    = threadIdx.x >> 5;
    const int l16  = lane & 15;
    const int hi   = lane >> 4;
    const int m0   = blockIdx.x * 128 + w * 16;

    const float* arow = A + (size_t)(m0 + l16) * N_NODES + hi * 8;

    // staging: thread -> (Bt row sn, 16-element half sp) of each 32-K half-plane
    const int sn = threadIdx.x >> 1;
    const int sp = threadIdx.x & 1;
    const __bf16* srow = Bt + (size_t)sn * N_NODES + sp * 16;
    __bf16* sdst[2] = { &bs[0][0][sn * BT_LDS_STRIDE + sp * 16],
                        &bs[1][0][sn * BT_LDS_STRIDE + sp * 16] };

    v8f acc[8] = {};

    // issue first chunk into buffer 0
    {
        const __bf16* s0 = srow;            // k = 0
#pragma unroll
        for (int kh = 0; kh < 2; ++kh) {
            copy16B_to_lds(s0 + kh * 32,     sdst[0] + kh * PLANE);
            copy16B_to_lds(s0 + kh * 32 + 8, sdst[0] + kh * PLANE + 8);
        }
    }

    int cur = 0;
    for (int it = 0; it < NITER; ++it) {
        wait_async_lds();        // this wave's stage of `cur` complete
        __syncthreads();         // all waves staged `cur`; all done reading `cur^1`

        if (it + 1 < NITER) {    // stage next chunk into the other buffer
            const __bf16* s0 = srow + (it + 1) * KCHUNK;
            __bf16* d0 = sdst[cur ^ 1];
#pragma unroll
            for (int kh = 0; kh < 2; ++kh) {
                copy16B_to_lds(s0 + kh * 32,     d0 + kh * PLANE);
                copy16B_to_lds(s0 + kh * 32 + 8, d0 + kh * PLANE + 8);
            }
        }

        const int k0 = it * KCHUNK;
#pragma unroll
        for (int kh = 0; kh < 2; ++kh) {
            v16bf af = cvt_a_frag(arow + k0 + kh * 32);
            const __bf16* pl = &bs[cur][kh][0];
#pragma unroll
            for (int nt = 0; nt < 8; ++nt) {
                BFrag b;
                const __bf16* bp = &pl[(nt * 16 + l16) * BT_LDS_STRIDE + hi * 8];
                b.q[0] = *(const v4f*)bp;
                b.q[1] = *(const v4f*)(bp + 16);
                acc[nt] = __builtin_amdgcn_wmma_f32_16x16x32_bf16(
                              false, af, false, b.v, (short)0, acc[nt], false, false);
            }
        }
        cur ^= 1;
    }

    if (mode == 0) {
#pragma unroll
        for (int nt = 0; nt < 8; ++nt) {
            int n = nt * 16 + l16;
            float bv = bias[n];
#pragma unroll
            for (int r = 0; r < 8; ++r) {
                float x = fmaxf(acc[nt][r] + bv, 0.0f);
                Hout[(size_t)(m0 + hi * 8 + r) * HDIM + n] = x;
            }
        }
    } else {
        float p[8] = {0.f, 0.f, 0.f, 0.f, 0.f, 0.f, 0.f, 0.f};
#pragma unroll
        for (int nt = 0; nt < 8; ++nt) {
            int n = nt * 16 + l16;
            float bv = bias[n];
            float wv = Wf[n];
#pragma unroll
            for (int r = 0; r < 8; ++r) {
                p[r] += fmaxf(acc[nt][r] + bv, 0.0f) * wv;
            }
        }
        float bfv = Bf[0];
#pragma unroll
        for (int r = 0; r < 8; ++r) {
            float s = p[r];
            s += __shfl_xor(s, 1, 16);
            s += __shfl_xor(s, 2, 16);
            s += __shfl_xor(s, 4, 16);
            s += __shfl_xor(s, 8, 16);
            if (l16 == 0) {
                Out[m0 + hi * 8 + r] = 1.0f / (1.0f + __expf(-(s + bfv)));
            }
        }
    }
}

// ---------------------------------------------------------------------------
// Host launcher
// ---------------------------------------------------------------------------
extern "C" void kernel_launch(void* const* d_in, const int* in_sizes, int n_in,
                              void* d_out, int out_size, void* d_ws, size_t ws_size,
                              hipStream_t stream) {
    const float* X  = (const float*)d_in[0];   // [8192][256]
    const float* A  = (const float*)d_in[1];   // [8192][8192]
    const float* W1 = (const float*)d_in[2];   // [256][128]
    const float* b1 = (const float*)d_in[3];   // [128]
    const float* W2 = (const float*)d_in[4];   // [128][128]
    const float* b2 = (const float*)d_in[5];   // [128]
    const float* Wf = (const float*)d_in[6];   // [128][1]
    const float* bf = (const float*)d_in[7];   // [1]
    float* out = (float*)d_out;                // [8192]

    char* ws = (char*)d_ws;
    size_t off = 0;
    __bf16* W1t = (__bf16*)(ws + off); off += (size_t)HDIM * F_IN * 2;
    __bf16* W2t = (__bf16*)(ws + off); off += (size_t)HDIM * HDIM * 2;
    __bf16* Bt1 = (__bf16*)(ws + off); off += (size_t)HDIM * N_NODES * 2;
    float*  h1  = (float*)(ws + off);  off += (size_t)N_NODES * HDIM * 4;
    __bf16* Bt2 = (__bf16*)(ws + off);

    hipLaunchKernelGGL(prep_weights, dim3(192), dim3(256), 0, stream, W1, W2, W1t, W2t);

    hipLaunchKernelGGL(small_gemm, dim3(512), dim3(256), 0, stream, X, W1t, Bt1, F_IN);

    hipLaunchKernelGGL(big_gemm, dim3(64), dim3(256), 0, stream,
                       A, Bt1, b1, h1,
                       (const float*)nullptr, (const float*)nullptr, (float*)nullptr, 0);

    hipLaunchKernelGGL(small_gemm, dim3(512), dim3(256), 0, stream, h1, W2t, Bt2, HDIM);

    hipLaunchKernelGGL(big_gemm, dim3(64), dim3(256), 0, stream,
                       A, Bt2, b2, (float*)nullptr, Wf, bf, out, 1);
}
